// BernNet_9543417332146
// MI455X (gfx1250) — compile-verified, matrix operand order checked by reading, not run on hardware
//
#include <hip/hip_runtime.h>
#include <hip/hip_bf16.h>

typedef __attribute__((ext_vector_type(16))) _Float16 v16h;
typedef __attribute__((ext_vector_type(8)))  float    v8f;

#define NNODES   50000
#define NEDGES   800000
#define FEAT     64
#define NHID     128
#define KPOLY    10
#define SCAN_BS  256

// ---------------------------------------------------------------------------
// Polynomial coefficients d_m of p(lambda) = 2^-k * sum_j C(k,j) T_j (1-l)^j (1+l)^{k-j}
// computed exactly in double from temp (T = relu(temp)).
// ---------------------------------------------------------------------------
__global__ void compute_coefs_kernel(const float* __restrict__ temp,
                                     float* __restrict__ dcoef) {
    if (threadIdx.x != 0 || blockIdx.x != 0) return;
    double C[KPOLY + 1][KPOLY + 1];
    for (int n = 0; n <= KPOLY; ++n) {
        C[n][0] = 1.0;
        for (int r = 1; r <= KPOLY; ++r) {
            if (r > n) { C[n][r] = 0.0; continue; }
            C[n][r] = C[n - 1][r - 1] + ((r <= n - 1) ? C[n - 1][r] : 0.0);
        }
    }
    for (int m = 0; m <= KPOLY; ++m) {
        double d = 0.0;
        for (int j = 0; j <= KPOLY; ++j) {
            double Tj = temp[j] > 0.0f ? (double)temp[j] : 0.0;
            double e = 0.0;
            int amax = (j < m) ? j : m;
            for (int a = 0; a <= amax; ++a) {
                int b = m - a;
                if (b > KPOLY - j) continue;
                double t = C[j][a] * C[KPOLY - j][b];
                e += (a & 1) ? -t : t;
            }
            d += C[KPOLY][j] * Tj * e;
        }
        dcoef[m] = (float)(d / 1024.0);  // 2^-k, k = 10
    }
}

// ---------------------------------------------------------------------------
// Degree histogram (src degree for dinv, dst degree for CSR rows)
// ---------------------------------------------------------------------------
__global__ void count_deg_kernel(const int* __restrict__ ei, int E,
                                 int* __restrict__ cnt_src, int* __restrict__ cnt_dst) {
    int e = blockIdx.x * blockDim.x + threadIdx.x;
    if (e < E) {
        atomicAdd(&cnt_src[ei[e]], 1);
        atomicAdd(&cnt_dst[ei[E + e]], 1);
    }
}

__global__ void dinv_kernel(const int* __restrict__ cnt_src,
                            float* __restrict__ dinv, int n) {
    int v = blockIdx.x * blockDim.x + threadIdx.x;
    if (v < n) {
        int c = cnt_src[v];
        dinv[v] = (c > 0) ? rsqrtf((float)c) : 0.0f;
    }
}

// ---------------------------------------------------------------------------
// Exclusive scan of cnt_dst -> row_ptr (3-pass block scan)
// ---------------------------------------------------------------------------
__global__ void scan1_kernel(const int* __restrict__ in, int* __restrict__ out,
                             int* __restrict__ bsums, int n) {
    __shared__ int sh[SCAN_BS];
    int i = blockIdx.x * SCAN_BS + threadIdx.x;
    int v = (i < n) ? in[i] : 0;
    sh[threadIdx.x] = v;
    __syncthreads();
    for (int off = 1; off < SCAN_BS; off <<= 1) {
        int t = (threadIdx.x >= off) ? sh[threadIdx.x - off] : 0;
        __syncthreads();
        sh[threadIdx.x] += t;
        __syncthreads();
    }
    int incl = sh[threadIdx.x];
    if (i < n) out[i] = incl - v;                       // exclusive within block
    if (threadIdx.x == SCAN_BS - 1) bsums[blockIdx.x] = incl;
}

__global__ void scan2_kernel(int* __restrict__ bsums, int nb) {
    __shared__ int sh[SCAN_BS];
    int v = (threadIdx.x < nb) ? bsums[threadIdx.x] : 0;
    sh[threadIdx.x] = v;
    __syncthreads();
    for (int off = 1; off < SCAN_BS; off <<= 1) {
        int t = (threadIdx.x >= off) ? sh[threadIdx.x - off] : 0;
        __syncthreads();
        sh[threadIdx.x] += t;
        __syncthreads();
    }
    if (threadIdx.x < nb) bsums[threadIdx.x] = sh[threadIdx.x] - v;  // exclusive
}

__global__ void scan3_kernel(int* __restrict__ out, const int* __restrict__ bsums,
                             int n, int total) {
    int i = blockIdx.x * SCAN_BS + threadIdx.x;
    if (i < n)       out[i] += bsums[blockIdx.x];
    else if (i == n) out[n]  = total;
}

__global__ void copy_int_kernel(const int* __restrict__ in, int* __restrict__ out, int n) {
    int i = blockIdx.x * blockDim.x + threadIdx.x;
    if (i < n) out[i] = in[i];
}

// ---------------------------------------------------------------------------
// CSR fill (atomic cursor), deterministic per-row sort, per-edge weights
// ---------------------------------------------------------------------------
__global__ void fill_csr_kernel(const int* __restrict__ ei, int E,
                                int* __restrict__ cursor, int* __restrict__ csr_src) {
    int e = blockIdx.x * blockDim.x + threadIdx.x;
    if (e < E) {
        int s = ei[e];
        int d = ei[E + e];
        int idx = atomicAdd(&cursor[d], 1);
        csr_src[idx] = s;
    }
}

// insertion-sort each row by src so the reduction order (and fp result) is
// independent of the atomic fill order -> bitwise-deterministic replays
__global__ void sort_rows_kernel(const int* __restrict__ rowp,
                                 int* __restrict__ csr_src, int n) {
    int v = blockIdx.x * blockDim.x + threadIdx.x;
    if (v >= n) return;
    int beg = rowp[v], end = rowp[v + 1];
    for (int i = beg + 1; i < end; ++i) {
        int key = csr_src[i];
        int j = i - 1;
        while (j >= beg && csr_src[j] > key) { csr_src[j + 1] = csr_src[j]; --j; }
        csr_src[j + 1] = key;
    }
}

__global__ void row_weights_kernel(const int* __restrict__ rowp,
                                   const int* __restrict__ csr_src,
                                   const float* __restrict__ dinv,
                                   float* __restrict__ csr_w, int n) {
    int v = blockIdx.x * blockDim.x + threadIdx.x;
    if (v >= n) return;
    float dv = dinv[v];
    int beg = rowp[v], end = rowp[v + 1];
    for (int e = beg; e < end; ++e) csr_w[e] = dv * dinv[csr_src[e]];
}

// ---------------------------------------------------------------------------
// MLP layer 1: h1 = relu(x @ W1 + b1), f16 output for layer-2 A operand.
// One wave32 per 16x16 tile; block = 8 waves = the 8 N-tiles of one M-tile.
// CDNA5 16-bit A/B fragment layout: element e (vgpr j = e/2, pair p = e&1)
// holds K = k0 + (j&3)*2 + (j>=4 ? 16 : 0) + half*8 + p, half = lane>>4.
// C/D: vgpr r -> row m0 + half*8 + r, col n0 + (lane&15).
// ---------------------------------------------------------------------------
__global__ void mlp1_kernel(const float* __restrict__ x,
                            const float* __restrict__ W1,
                            const float* __restrict__ b1,
                            _Float16* __restrict__ h1) {
    int wave = threadIdx.x >> 5;          // n-tile 0..7
    int lane = threadIdx.x & 31;
    int mtile = blockIdx.x;               // 0..3124
    int half = lane >> 4, lr = lane & 15;
    int M = mtile * 16 + lr;
    int N = wave * 16 + lr;

    v8f acc = {};
    #pragma unroll
    for (int k0 = 0; k0 < FEAT; k0 += 32) {
        v16h a, b;
        #pragma unroll
        for (int j = 0; j < 8; ++j) {
            int kk = k0 + ((j & 3) * 2) + ((j >= 4) ? 16 : 0) + half * 8;
            a[2 * j]     = (_Float16)x[M * FEAT + kk];
            a[2 * j + 1] = (_Float16)x[M * FEAT + kk + 1];
            b[2 * j]     = (_Float16)W1[kk * NHID + N];
            b[2 * j + 1] = (_Float16)W1[(kk + 1) * NHID + N];
        }
        acc = __builtin_amdgcn_wmma_f32_16x16x32_f16(false, a, false, b,
                                                     (short)0, acc, false, false);
    }
    float bias = b1[N];
    #pragma unroll
    for (int r = 0; r < 8; ++r) {
        int Mr = mtile * 16 + half * 8 + r;
        float v = acc[r] + bias;
        h1[Mr * NHID + N] = (_Float16)(v > 0.0f ? v : 0.0f);
    }
}

// MLP layer 2: h = h1 @ W2 + b2 (f32 out). Block = 4 waves = 4 N-tiles.
__global__ void mlp2_kernel(const _Float16* __restrict__ h1,
                            const float* __restrict__ W2,
                            const float* __restrict__ b2,
                            float* __restrict__ h) {
    int wave = threadIdx.x >> 5;          // n-tile 0..3
    int lane = threadIdx.x & 31;
    int mtile = blockIdx.x;
    int half = lane >> 4, lr = lane & 15;
    int M = mtile * 16 + lr;
    int N = wave * 16 + lr;

    v8f acc = {};
    #pragma unroll
    for (int k0 = 0; k0 < NHID; k0 += 32) {
        v16h a, b;
        #pragma unroll
        for (int j = 0; j < 8; ++j) {
            int kk = k0 + ((j & 3) * 2) + ((j >= 4) ? 16 : 0) + half * 8;
            a[2 * j]     = h1[M * NHID + kk];
            a[2 * j + 1] = h1[M * NHID + kk + 1];
            b[2 * j]     = (_Float16)W2[kk * FEAT + N];
            b[2 * j + 1] = (_Float16)W2[(kk + 1) * FEAT + N];
        }
        acc = __builtin_amdgcn_wmma_f32_16x16x32_f16(false, a, false, b,
                                                     (short)0, acc, false, false);
    }
    float bias = b2[N];
    #pragma unroll
    for (int r = 0; r < 8; ++r) {
        int Mr = mtile * 16 + half * 8 + r;
        h[Mr * FEAT + N] = acc[r] + bias;
    }
}

// q = dcoef[K] * h  (Horner init)
__global__ void scale_kernel(const float* __restrict__ h,
                             const float* __restrict__ dcoef,
                             float* __restrict__ q, int n) {
    int i = blockIdx.x * blockDim.x + threadIdx.x;
    if (i < n) q[i] = dcoef[KPOLY] * h[i];
}

// ---------------------------------------------------------------------------
// Horner step: q_out[v,:] = dcoef[m] * h[v,:] + sum_{e in row v} w_e * q_in[src_e,:]
// One wave32 per dst row; lane l holds features {2l, 2l+1} as float2.
// Edges broadcast across the wave with __shfl; q gathers are coalesced 256B rows
// that stay resident in the 192MB L2 (q is 12.8MB).
// ---------------------------------------------------------------------------
__global__ void propagate_kernel(const int* __restrict__ rowp,
                                 const int* __restrict__ csr_src,
                                 const float* __restrict__ csr_w,
                                 const float* __restrict__ qin,
                                 const float* __restrict__ h,
                                 const float* __restrict__ dcoef, int m,
                                 float* __restrict__ qout, int n) {
    int wid  = (blockIdx.x * blockDim.x + threadIdx.x) >> 5;  // row = wave id
    int lane = threadIdx.x & 31;
    if (wid >= n) return;

    int beg = rowp[wid], end = rowp[wid + 1];
    const float2* q2 = (const float2*)qin;
    float2 acc = {0.0f, 0.0f};

    for (int e = beg; e < end; e += 32) {
        int cnt = end - e;
        if (cnt > 32) cnt = 32;
        int   s  = 0;
        float ww = 0.0f;
        if (lane < cnt) { s = csr_src[e + lane]; ww = csr_w[e + lane]; }
        for (int j = 0; j < cnt; ++j) {
            int   ss  = __shfl(s, j, 32);
            float www = __shfl(ww, j, 32);
            float2 v = q2[ss * 32 + lane];
            acc.x += www * v.x;
            acc.y += www * v.y;
        }
    }
    float dm = dcoef[m];
    float2 hv = ((const float2*)h)[wid * 32 + lane];
    float2 o;
    o.x = dm * hv.x + acc.x;
    o.y = dm * hv.y + acc.y;
    ((float2*)qout)[wid * 32 + lane] = o;
}

// ---------------------------------------------------------------------------
extern "C" void kernel_launch(void* const* d_in, const int* in_sizes, int n_in,
                              void* d_out, int out_size, void* d_ws, size_t ws_size,
                              hipStream_t stream) {
    const float* x    = (const float*)d_in[0];
    const int*   ei   = (const int*)d_in[1];     // [2, E] row-major (int32)
    const float* W1   = (const float*)d_in[2];
    const float* b1   = (const float*)d_in[3];
    const float* W2   = (const float*)d_in[4];
    const float* b2   = (const float*)d_in[5];
    const float* temp = (const float*)d_in[6];

    const int n = in_sizes[0] / FEAT;            // 50000
    const int E = in_sizes[1] / 2;               // 800000

    // ---- workspace layout (256B aligned slices) ----
    char*  w   = (char*)d_ws;
    size_t off = 0;
    auto take = [&](size_t bytes) -> void* {
        off = (off + 255) & ~(size_t)255;
        void* p = w + off;
        off += bytes;
        return p;
    };
    float*     hbuf    = (float*)    take((size_t)n * FEAT * 4);
    _Float16*  h1buf   = (_Float16*) take((size_t)n * NHID * 2);
    float*     q0      = (float*)    take((size_t)n * FEAT * 4);
    float*     q1      = (float*)    take((size_t)n * FEAT * 4);
    int*       cnt_src = (int*)      take((size_t)n * 4);
    int*       cnt_dst = (int*)      take((size_t)n * 4);
    float*     dinv    = (float*)    take((size_t)n * 4);
    int*       rowp    = (int*)      take((size_t)(n + 1) * 4);
    int*       cursor  = (int*)      take((size_t)n * 4);
    int*       bsums   = (int*)      take((size_t)SCAN_BS * 4);
    float*     dcoef   = (float*)    take(64);
    int*       csr_src = (int*)      take((size_t)E * 4);
    float*     csr_w   = (float*)    take((size_t)E * 4);
    (void)ws_size;

    const int nb_n = (n + 255) / 256;            // 196
    const int nb_e = (E + 255) / 256;            // 3125
    const int nb_s3 = (n + 1 + SCAN_BS - 1) / SCAN_BS;

    // ---- graph preprocessing: degrees, dinv, CSR by dst ----
    hipMemsetAsync(cnt_src, 0, (size_t)n * 4, stream);
    hipMemsetAsync(cnt_dst, 0, (size_t)n * 4, stream);
    count_deg_kernel<<<nb_e, 256, 0, stream>>>(ei, E, cnt_src, cnt_dst);
    dinv_kernel<<<nb_n, 256, 0, stream>>>(cnt_src, dinv, n);

    scan1_kernel<<<nb_n, SCAN_BS, 0, stream>>>(cnt_dst, rowp, bsums, n);
    scan2_kernel<<<1, SCAN_BS, 0, stream>>>(bsums, nb_n);
    scan3_kernel<<<nb_s3, SCAN_BS, 0, stream>>>(rowp, bsums, n, E);

    copy_int_kernel<<<nb_n, 256, 0, stream>>>(rowp, cursor, n);
    fill_csr_kernel<<<nb_e, 256, 0, stream>>>(ei, E, cursor, csr_src);
    sort_rows_kernel<<<nb_n, 256, 0, stream>>>(rowp, csr_src, n);
    row_weights_kernel<<<nb_n, 256, 0, stream>>>(rowp, csr_src, dinv, csr_w, n);

    // ---- Bernstein -> monomial coefficients from temp ----
    compute_coefs_kernel<<<1, 32, 0, stream>>>(temp, dcoef);

    // ---- MLP head via WMMA ----
    mlp1_kernel<<<n / 16, 256, 0, stream>>>(x, W1, b1, h1buf);   // 3125 blocks, 8 waves
    mlp2_kernel<<<n / 16, 128, 0, stream>>>(h1buf, W2, b2, hbuf);

    // ---- Horner evaluation: out = sum_m dcoef[m] A^m h, 10 matvecs total ----
    const int nelem = n * FEAT;
    scale_kernel<<<(nelem + 255) / 256, 256, 0, stream>>>(hbuf, dcoef, q0, nelem);

    float* qa = q0;
    float* qb = q1;
    const int prop_blocks = (n * 32 + 255) / 256;   // one wave32 per row
    for (int m = KPOLY - 1; m >= 1; --m) {
        propagate_kernel<<<prop_blocks, 256, 0, stream>>>(rowp, csr_src, csr_w,
                                                          qa, hbuf, dcoef, m, qb, n);
        float* t = qa; qa = qb; qb = t;
    }
    propagate_kernel<<<prop_blocks, 256, 0, stream>>>(rowp, csr_src, csr_w,
                                                      qa, hbuf, dcoef, 0,
                                                      (float*)d_out, n);
}
